// SpineSegmentationNet_77025943486588
// MI455X (gfx1250) — matrix operand order repaired and checked by reading, not compiled, not run on hardware
//
#include <hip/hip_runtime.h>

// ---------------------------------------------------------------------------
// SpineSegmentationNet on MI455X (gfx1250, wave32).
// KNN selection: per-wave owner-rescan argmin in LDS (barrier-free).
// MLP GEMMs: v_wmma_f32_16x16x32_f16 (f16 in, f32 accumulate), with
// fragment-major swizzled weight staging so every B fragment is a single
// 32-byte LDS read (2x ds_load_b128) and A fragments are 2x ds_load_b128.
// Weights staged once per multi-wave block and shared across waves.
// ---------------------------------------------------------------------------

typedef __attribute__((ext_vector_type(16))) _Float16 v16h;
typedef __attribute__((ext_vector_type(8)))  _Float16 v8h;
typedef __attribute__((ext_vector_type(8)))  float    v8f;

#define WMMA_F32_F16(a, b, c) \
  __builtin_amdgcn_wmma_f32_16x16x32_f16(false, (a), false, (b), (short)0, (c), false, false)

static constexpr int BB = 8;
static constexpr int NN = 16384;
static constexpr int S1 = 512;
static constexpr int S2 = 128;
static constexpr int GG = 32;
#define INF 3.0e38f

// ---- WMMA fragment access (layouts per CDNA5 ISA 7.12.2) -------
// A: 16x32 f16 tile, row-major in LDS, lda in halves (multiple of 8).
// Per lane: halves 0..7 = K[8g..8g+7], halves 8..15 = K[16+8g..16+8g+7].
__device__ __forceinline__ v16h load_frag_a(const _Float16* A, int lda, int lane) {
  const int m = lane & 15, g = lane >> 4;
  const _Float16* p = A + m * lda + (g << 3);
  v8h lo = *(const v8h*)(p);
  v8h hi = *(const v8h*)(p + 16);
  return __builtin_shufflevector(lo, hi, 0, 1, 2, 3, 4, 5, 6, 7,
                                 8, 9, 10, 11, 12, 13, 14, 15);
}

// Swizzled B staging: tile (kt,nt) of a K x N matrix; lane (g*16+nn) holds
// element (k = kt*32 + (h<8 ? 8g+h : 16+8g+h-8), n = nt*16 + nn) at slot
// ((kt*NT + nt)*32 + lane)*16 + h  -> contiguous 32 bytes per lane.
__device__ __forceinline__ void stage_b_swz(_Float16* dst, const float* __restrict__ W,
                                            int Kpad, int Ncols, int Kreal,
                                            int t, int nthreads) {
  const int NT = Ncols >> 4;
  for (int i = t; i < Kpad * Ncols; i += nthreads) {
    const int k = i / Ncols, n = i % Ncols;
    const int kt = k >> 5, kin = k & 31;
    const int g = (kin >> 3) & 1, h = (kin & 7) + ((kin >> 4) << 3);
    const int nt = n >> 4, nn = n & 15;
    const int lane = (g << 4) + nn;
    const float v = (k < Kreal) ? W[k * Ncols + n] : 0.f;
    dst[(((kt * NT) + nt) * 32 + lane) * 16 + h] = (_Float16)v;
  }
}

__device__ __forceinline__ v16h load_frag_b_swz(const _Float16* Wswz, int NT,
                                                int kt, int nt, int lane) {
  return *(const v16h*)(Wswz + (((kt * NT) + nt) * 32 + lane) * 16);
}

// Wave-wide lexicographic argmin reduce over (value, index).
__device__ __forceinline__ void wave_min_pair(float& v, int& idx) {
#pragma unroll
  for (int off = 16; off > 0; off >>= 1) {
    const float ov = __shfl_xor(v, off, 32);
    const int   oi = __shfl_xor(idx, off, 32);
    if (ov < v || (ov == v && oi < idx)) { v = ov; idx = oi; }
  }
}

// ---------------------------------------------------------------------------
// 1) samp1 = x[:, sample_idx1]
__global__ void samp1_kernel(const float* __restrict__ x, const int* __restrict__ sidx1,
                             float* __restrict__ samp1) {
  const int i = blockIdx.x * blockDim.x + threadIdx.x;
  if (i >= BB * S1) return;
  const int b = i >> 9;
  const int s = i & (S1 - 1);
  const int n = sidx1[s];
  const float* p = x + ((size_t)b * NN + n) * 3;
  float* o = samp1 + (size_t)i * 3;
  o[0] = p[0]; o[1] = p[1]; o[2] = p[2];
}

// ---------------------------------------------------------------------------
// 2) knn1: for each (b,q) find 32 nearest of 16384 points (3-D).
//    Chunked: 4 x 4096 -> 128 candidates -> final top-32 merge.
__global__ __launch_bounds__(32) void knn1_kernel(const float* __restrict__ x,
                                                  const float* __restrict__ samp1,
                                                  int* __restrict__ idx1) {
  const int blk = blockIdx.x;              // b*S1 + q
  const int b = blk >> 9;
  const int lane = threadIdx.x;
  __shared__ float sd[4096];
  __shared__ float cd[128];
  __shared__ int   ci[128];
  const float* qp = samp1 + (size_t)blk * 3;
  const float qx = qp[0], qy = qp[1], qz = qp[2];
  const float* xb = x + (size_t)b * NN * 3;
  int cbase = 0;
  for (int c0 = 0; c0 < NN; c0 += 4096) {
    for (int r = lane; r < 4096; r += 32) {
      const float* p = xb + (size_t)(c0 + r) * 3;
      const float dx = p[0] - qx, dy = p[1] - qy, dz = p[2] - qz;
      sd[r] = dx * dx + dy * dy + dz * dz;   // lane-owned entries only
      if (c0 + 4096 < NN)                    // pull next chunk toward L2/L0
        __builtin_prefetch(xb + (size_t)(c0 + 4096 + r) * 3, 0, 1);
    }
    float lv = INF; int li = 0x7fffffff;
    for (int r = lane; r < 4096; r += 32) { const float d = sd[r]; if (d < lv) { lv = d; li = r; } }
    for (int k = 0; k < 32; ++k) {
      float v = lv; int idx = li;
      wave_min_pair(v, idx);
      if (lane == 0) { cd[cbase + k] = v; ci[cbase + k] = c0 + idx; }
      if (lane == (idx & 31)) {            // owner invalidates + rescans its stripe
        sd[idx] = INF;
        lv = INF; li = 0x7fffffff;
        for (int r = lane; r < 4096; r += 32) { const float d = sd[r]; if (d < lv) { lv = d; li = r; } }
      }
    }
    cbase += 32;
  }
  __syncthreads();                         // cd/ci written by lane 0, read by all
  float lv = INF; int li = 0x7fffffff;
  for (int r = lane; r < 128; r += 32) { const float d = cd[r]; if (d < lv) { lv = d; li = r; } }
  int* out = idx1 + (size_t)blk * GG;
  for (int k = 0; k < GG; ++k) {
    float v = lv; int idx = li;
    wave_min_pair(v, idx);
    if (lane == 0) out[k] = ci[idx];
    if (lane == (idx & 31)) {
      cd[idx] = INF;
      lv = INF; li = 0x7fffffff;
      for (int r = lane; r < 128; r += 32) { const float d = cd[r]; if (d < lv) { lv = d; li = r; } }
    }
  }
}

// ---------------------------------------------------------------------------
// 3) sa1: gather 32 neighbors (xyz) -> relu(A@W1+b1)@W2+b2 -> max over 32.
//    4 waves per block; swizzled W1/W2 staged once and shared.
__global__ __launch_bounds__(128) void sa1_kernel(const float* __restrict__ x,
                                                  const int* __restrict__ idx1,
                                                  const float* __restrict__ W1,
                                                  const float* __restrict__ b1v,
                                                  const float* __restrict__ W2,
                                                  const float* __restrict__ b2v,
                                                  float* __restrict__ feat1) {
  const int t = threadIdx.x, wv = t >> 5, lane = t & 31;
  const int blk = blockIdx.x * 4 + wv;     // b*S1 + s
  const int b = blk >> 9;
  __shared__ __align__(32) _Float16 W1s[32 * 64];    // swz: KT=1, NT=4 (K 3->32 pad)
  __shared__ __align__(32) _Float16 W2s[64 * 64];    // swz: KT=2, NT=4
  __shared__ __align__(32) _Float16 Ah[4][32 * 32];
  __shared__ __align__(32) _Float16 Hh[4][32 * 64];
  __shared__ float bias1[64], bias2[64];
  stage_b_swz(W1s, W1, 32, 64, 3, t, 128);
  stage_b_swz(W2s, W2, 64, 64, 64, t, 128);
  if (t < 64) { bias1[t] = b1v[t]; bias2[t] = b2v[t]; }
  {
    const int n = idx1[(size_t)blk * GG + lane];
    const float* p = x + ((size_t)b * NN + n) * 3;
    _Float16* ar = &Ah[wv][lane * 32];
#pragma unroll
    for (int k = 0; k < 32; ++k) ar[k] = (_Float16)0.f;
    ar[0] = (_Float16)p[0]; ar[1] = (_Float16)p[1]; ar[2] = (_Float16)p[2];
  }
  __syncthreads();
  const int m = lane & 15, g = lane >> 4;
  // GEMM1: (32x32)@(32x64) + bias, relu
  for (int nt = 0; nt < 4; ++nt)
    for (int mt = 0; mt < 2; ++mt) {
      v8f c = {};
      v16h a  = load_frag_a(&Ah[wv][mt * 16 * 32], 32, lane);
      v16h bb = load_frag_b_swz(W1s, 4, 0, nt, lane);
      c = WMMA_F32_F16(a, bb, c);
      const float bn = bias1[nt * 16 + m];
#pragma unroll
      for (int r = 0; r < 8; ++r) {
        float v = c[r] + bn; v = v > 0.f ? v : 0.f;
        Hh[wv][(mt * 16 + r + 8 * g) * 64 + nt * 16 + m] = (_Float16)v;
      }
    }
  __syncthreads();
  // GEMM2: (32x64)@(64x64) + bias, column max over 32 rows
  for (int nt = 0; nt < 4; ++nt) {
    float mx = -INF;
    for (int mt = 0; mt < 2; ++mt) {
      v8f c = {};
      for (int kt = 0; kt < 2; ++kt) {
        v16h a  = load_frag_a(&Hh[wv][mt * 16 * 64 + kt * 32], 64, lane);
        v16h bb = load_frag_b_swz(W2s, 4, kt, nt, lane);
        c = WMMA_F32_F16(a, bb, c);
      }
      const float bn = bias2[nt * 16 + m];
#pragma unroll
      for (int r = 0; r < 8; ++r) { const float v = c[r] + bn; mx = v > mx ? v : mx; }
    }
    const float o = __shfl_xor(mx, 16, 32);
    mx = o > mx ? o : mx;
    if (g == 0) feat1[(size_t)blk * 64 + nt * 16 + m] = mx;
  }
}

// ---------------------------------------------------------------------------
// 4) knn2: queries feat1[:, sample_idx2] (64-D), refs feat1 (512), top-32.
__global__ __launch_bounds__(32) void knn2_kernel(const float* __restrict__ feat1,
                                                  const int* __restrict__ sidx2,
                                                  int* __restrict__ idx2) {
  const int blk = blockIdx.x;              // b*S2 + j
  const int b = blk >> 7, j = blk & (S2 - 1);
  const int lane = threadIdx.x;
  __shared__ float qv[64];
  __shared__ float sd[S1];
  const float* fb = feat1 + (size_t)b * S1 * 64;
  const int qrow = sidx2[j];
  for (int c = lane; c < 64; c += 32) qv[c] = fb[(size_t)qrow * 64 + c];
  __syncthreads();
  for (int r = lane; r < S1; r += 32) {
    const float* p = fb + (size_t)r * 64;
    float acc = 0.f;
#pragma unroll 8
    for (int c = 0; c < 64; ++c) { const float d = p[c] - qv[c]; acc += d * d; }
    sd[r] = acc;
  }
  float lv = INF; int li = 0x7fffffff;
  for (int r = lane; r < S1; r += 32) { const float d = sd[r]; if (d < lv) { lv = d; li = r; } }
  int* out = idx2 + (size_t)blk * GG;
  for (int k = 0; k < GG; ++k) {
    float v = lv; int idx = li;
    wave_min_pair(v, idx);
    if (lane == 0) out[k] = idx;
    if (lane == (idx & 31)) {
      sd[idx] = INF;
      lv = INF; li = 0x7fffffff;
      for (int r = lane; r < S1; r += 32) { const float d = sd[r]; if (d < lv) { lv = d; li = r; } }
    }
  }
}

// ---------------------------------------------------------------------------
// 5) sa2: gather 32 neighbor feat1 rows (64) -> 128 -> 128, max over 32.
//    1 wave per block (LDS-bound: ~62 KB resident tiles).
__global__ __launch_bounds__(32) void sa2_kernel(const float* __restrict__ feat1,
                                                 const int* __restrict__ idx2,
                                                 const float* __restrict__ W1,
                                                 const float* __restrict__ b1v,
                                                 const float* __restrict__ W2,
                                                 const float* __restrict__ b2v,
                                                 float* __restrict__ feat2) {
  const int blk = blockIdx.x;              // b*S2 + j
  const int b = blk >> 7;
  const int lane = threadIdx.x;
  __shared__ __align__(32) _Float16 Ah[32 * 64];       // 4 KB
  __shared__ __align__(32) _Float16 W1s[64 * 128];     // 16 KB swz: KT=2, NT=8
  __shared__ __align__(32) _Float16 Hh[32 * 128];      // 8 KB
  __shared__ __align__(32) _Float16 W2s[128 * 128];    // 32 KB swz: KT=4, NT=8
  __shared__ float bias1[128], bias2[128];
  {
    const int n = idx2[(size_t)blk * GG + lane];
    const float* p = feat1 + ((size_t)b * S1 + n) * 64;
    for (int c = 0; c < 64; ++c) Ah[lane * 64 + c] = (_Float16)p[c];
  }
  stage_b_swz(W1s, W1, 64, 128, 64, lane, 32);
  stage_b_swz(W2s, W2, 128, 128, 128, lane, 32);
  for (int i = lane; i < 128; i += 32) { bias1[i] = b1v[i]; bias2[i] = b2v[i]; }
  __syncthreads();
  const int m = lane & 15, g = lane >> 4;
  for (int nt = 0; nt < 8; ++nt)
    for (int mt = 0; mt < 2; ++mt) {
      v8f c = {};
      for (int kt = 0; kt < 2; ++kt) {
        v16h a  = load_frag_a(Ah + mt * 16 * 64 + kt * 32, 64, lane);
        v16h bb = load_frag_b_swz(W1s, 8, kt, nt, lane);
        c = WMMA_F32_F16(a, bb, c);
      }
      const float bn = bias1[nt * 16 + m];
#pragma unroll
      for (int r = 0; r < 8; ++r) {
        float v = c[r] + bn; v = v > 0.f ? v : 0.f;
        Hh[(mt * 16 + r + 8 * g) * 128 + nt * 16 + m] = (_Float16)v;
      }
    }
  __syncthreads();
  for (int nt = 0; nt < 8; ++nt) {
    float mx = -INF;
    for (int mt = 0; mt < 2; ++mt) {
      v8f c = {};
      for (int kt = 0; kt < 4; ++kt) {
        v16h a  = load_frag_a(Hh + mt * 16 * 128 + kt * 32, 128, lane);
        v16h bb = load_frag_b_swz(W2s, 8, kt, nt, lane);
        c = WMMA_F32_F16(a, bb, c);
      }
      const float bn = bias2[nt * 16 + m];
#pragma unroll
      for (int r = 0; r < 8; ++r) { const float v = c[r] + bn; mx = v > mx ? v : mx; }
    }
    const float o = __shfl_xor(mx, 16, 32);
    mx = o > mx ? o : mx;
    if (g == 0) feat2[(size_t)blk * 128 + nt * 16 + m] = mx;
  }
}

// ---------------------------------------------------------------------------
// 6) interp1: knn3(samp1 -> coords2, k=3), mean-gather feat2 -> (B,512,128)
__global__ __launch_bounds__(128) void interp1_kernel(const float* __restrict__ samp1,
                                                      const int* __restrict__ sidx2,
                                                      const float* __restrict__ feat2,
                                                      float* __restrict__ interp1) {
  const int blk = blockIdx.x;              // B*4 blocks
  const int b = blk >> 2, qt = blk & 3;
  const int t = threadIdx.x;
  __shared__ float refs[S2 * 3];
  __shared__ int   sidx[S2 * 3];
  const float* s1b = samp1 + (size_t)b * S1 * 3;
  {
    const int rj = sidx2[t];               // coords2[t] = samp1[b][sidx2[t]]
    refs[t * 3 + 0] = s1b[rj * 3 + 0];
    refs[t * 3 + 1] = s1b[rj * 3 + 1];
    refs[t * 3 + 2] = s1b[rj * 3 + 2];
  }
  __syncthreads();
  const int q = qt * 128 + t;
  const float* qp = s1b + (size_t)q * 3;
  const float qx = qp[0], qy = qp[1], qz = qp[2];
  float d0 = INF, d1 = INF, d2 = INF; int i0 = 0, i1 = 0, i2 = 0;
  for (int rj = 0; rj < S2; ++rj) {
    const float dx = refs[rj * 3] - qx, dy = refs[rj * 3 + 1] - qy, dz = refs[rj * 3 + 2] - qz;
    const float d = dx * dx + dy * dy + dz * dz;
    if (d < d0)      { d2 = d1; i2 = i1; d1 = d0; i1 = i0; d0 = d; i0 = rj; }
    else if (d < d1) { d2 = d1; i2 = i1; d1 = d; i1 = rj; }
    else if (d < d2) { d2 = d; i2 = rj; }
  }
  sidx[t * 3] = i0; sidx[t * 3 + 1] = i1; sidx[t * 3 + 2] = i2;
  __syncthreads();
  const float* f2b = feat2 + (size_t)b * S2 * 128;
  for (int qq = 0; qq < 128; ++qq) {
    const int a0 = sidx[qq * 3], a1 = sidx[qq * 3 + 1], a2 = sidx[qq * 3 + 2];
    const int c = t;
    const float v = (f2b[(size_t)a0 * 128 + c] + f2b[(size_t)a1 * 128 + c] +
                     f2b[(size_t)a2 * 128 + c]) * (1.f / 3.f);
    interp1[((size_t)b * S1 + qt * 128 + qq) * 128 + c] = v;
  }
}

// ---------------------------------------------------------------------------
// 7) fp1: h1 = relu(concat(feat1, interp1) @ W1 + b1) @ W2 + b2  (192->64->64)
//    2 waves per block sharing swizzled weights.
__global__ __launch_bounds__(64) void fp1_kernel(const float* __restrict__ feat1,
                                                 const float* __restrict__ interp1,
                                                 const float* __restrict__ W1,
                                                 const float* __restrict__ b1v,
                                                 const float* __restrict__ W2,
                                                 const float* __restrict__ b2v,
                                                 float* __restrict__ h1) {
  const int t = threadIdx.x, wv = t >> 5, lane = t & 31;
  const size_t R0 = ((size_t)blockIdx.x * 2 + wv) * 16;   // row base over B*S1
  __shared__ __align__(32) _Float16 W1s[192 * 64];   // swz: KT=6, NT=4
  __shared__ __align__(32) _Float16 W2s[64 * 64];    // swz: KT=2, NT=4
  __shared__ __align__(32) _Float16 Ah[2][16 * 192];
  __shared__ __align__(32) _Float16 Hh[2][16 * 64];
  __shared__ float bias1[64], bias2[64];
  stage_b_swz(W1s, W1, 192, 64, 192, t, 64);
  stage_b_swz(W2s, W2, 64, 64, 64, t, 64);
  bias1[t] = b1v[t];
  bias2[t] = b2v[t];
  for (int i = lane; i < 16 * 192; i += 32) {
    const int r = i / 192, c = i % 192;
    const size_t R = R0 + r;
    const float v = (c < 64) ? feat1[R * 64 + c] : interp1[R * 128 + (c - 64)];
    Ah[wv][i] = (_Float16)v;
  }
  __syncthreads();
  const int m = lane & 15, g = lane >> 4;
  for (int nt = 0; nt < 4; ++nt) {
    v8f c = {};
    for (int kt = 0; kt < 6; ++kt) {
      v16h a  = load_frag_a(&Ah[wv][kt * 32], 192, lane);
      v16h bb = load_frag_b_swz(W1s, 4, kt, nt, lane);
      c = WMMA_F32_F16(a, bb, c);
    }
    const float bn = bias1[nt * 16 + m];
#pragma unroll
    for (int r = 0; r < 8; ++r) {
      float v = c[r] + bn; v = v > 0.f ? v : 0.f;
      Hh[wv][(r + 8 * g) * 64 + nt * 16 + m] = (_Float16)v;
    }
  }
  __syncthreads();
  for (int nt = 0; nt < 4; ++nt) {
    v8f c = {};
    for (int kt = 0; kt < 2; ++kt) {
      v16h a  = load_frag_a(&Hh[wv][kt * 32], 64, lane);
      v16h bb = load_frag_b_swz(W2s, 4, kt, nt, lane);
      c = WMMA_F32_F16(a, bb, c);
    }
    const float bn = bias2[nt * 16 + m];
#pragma unroll
    for (int r = 0; r < 8; ++r)
      h1[(R0 + r + 8 * g) * 64 + nt * 16 + m] = c[r] + bn;
  }
}

// ---------------------------------------------------------------------------
// 8) interp2: knn(x -> samp1, k=3), mean-gather h1 -> (B,16384,64)
__global__ __launch_bounds__(256) void interp2_kernel(const float* __restrict__ x,
                                                      const float* __restrict__ samp1,
                                                      const float* __restrict__ h1,
                                                      float* __restrict__ interp2) {
  const int blk = blockIdx.x;              // B*64 blocks
  const int b = blk >> 6, qt = blk & 63;
  const int t = threadIdx.x;
  __shared__ float refs[S1 * 3];
  __shared__ int   sidx[256 * 3];
  const float* s1b = samp1 + (size_t)b * S1 * 3;
  for (int i = t; i < S1 * 3; i += 256) refs[i] = s1b[i];
  __syncthreads();
  const int q = qt * 256 + t;
  const float* qp = x + ((size_t)b * NN + q) * 3;
  const float qx = qp[0], qy = qp[1], qz = qp[2];
  float d0 = INF, d1 = INF, d2 = INF; int i0 = 0, i1 = 0, i2 = 0;
  for (int rj = 0; rj < S1; ++rj) {
    const float dx = refs[rj * 3] - qx, dy = refs[rj * 3 + 1] - qy, dz = refs[rj * 3 + 2] - qz;
    const float d = dx * dx + dy * dy + dz * dz;
    if (d < d0)      { d2 = d1; i2 = i1; d1 = d0; i1 = i0; d0 = d; i0 = rj; }
    else if (d < d1) { d2 = d1; i2 = i1; d1 = d; i1 = rj; }
    else if (d < d2) { d2 = d; i2 = rj; }
  }
  sidx[t * 3] = i0; sidx[t * 3 + 1] = i1; sidx[t * 3 + 2] = i2;
  __syncthreads();
  const float* h1b = h1 + (size_t)b * S1 * 64;
  for (int pass = 0; pass < 64; ++pass) {
    const int qq = pass * 4 + (t >> 6);
    const int c = t & 63;
    const int a0 = sidx[qq * 3], a1 = sidx[qq * 3 + 1], a2 = sidx[qq * 3 + 2];
    const float v = (h1b[(size_t)a0 * 64 + c] + h1b[(size_t)a1 * 64 + c] +
                     h1b[(size_t)a2 * 64 + c]) * (1.f / 3.f);
    interp2[((size_t)b * NN + qt * 256 + qq) * 64 + c] = v;
  }
}

// ---------------------------------------------------------------------------
// 9) fp2 + fc + sigmoid: concat(x, interp2) (67) -> 32 -> 32 -> 2
//    4 waves per block sharing swizzled weights.
__global__ __launch_bounds__(128) void fp2_kernel(const float* __restrict__ x,
                                                  const float* __restrict__ interp2,
                                                  const float* __restrict__ W1,
                                                  const float* __restrict__ b1v,
                                                  const float* __restrict__ W2,
                                                  const float* __restrict__ b2v,
                                                  const float* __restrict__ fcW,
                                                  const float* __restrict__ fcb,
                                                  float* __restrict__ out) {
  const int t = threadIdx.x, wv = t >> 5, lane = t & 31;
  const size_t R0 = ((size_t)blockIdx.x * 4 + wv) * 16;   // row base over B*N
  __shared__ __align__(32) _Float16 W1s[96 * 32];    // swz: KT=3, NT=2 (K 67->96 pad)
  __shared__ __align__(32) _Float16 W2s[32 * 32];    // swz: KT=1, NT=2
  __shared__ __align__(32) _Float16 Ah[4][16 * 96];
  __shared__ __align__(32) _Float16 Hh[4][16 * 32];
  __shared__ float h2s[4][16 * 32];
  __shared__ float bias1[32], bias2[32], fw[64], fb2[2];
  stage_b_swz(W1s, W1, 96, 32, 67, t, 128);
  stage_b_swz(W2s, W2, 32, 32, 32, t, 128);
  if (t < 32) { bias1[t] = b1v[t]; bias2[t] = b2v[t]; }
  if (t < 64) fw[t] = fcW[t];
  if (t < 2) fb2[t] = fcb[t];
  for (int i = lane; i < 16 * 96; i += 32) {
    const int r = i / 96, c = i % 96;
    const size_t R = R0 + r;
    float v;
    if (c < 3)       v = x[R * 3 + c];
    else if (c < 67) v = interp2[R * 64 + (c - 3)];
    else             v = 0.f;
    Ah[wv][i] = (_Float16)v;
  }
  __syncthreads();
  const int m = lane & 15, g = lane >> 4;
  for (int nt = 0; nt < 2; ++nt) {
    v8f c = {};
    for (int kt = 0; kt < 3; ++kt) {
      v16h a  = load_frag_a(&Ah[wv][kt * 32], 96, lane);
      v16h bb = load_frag_b_swz(W1s, 2, kt, nt, lane);
      c = WMMA_F32_F16(a, bb, c);
    }
    const float bn = bias1[nt * 16 + m];
#pragma unroll
    for (int r = 0; r < 8; ++r) {
      float v = c[r] + bn; v = v > 0.f ? v : 0.f;
      Hh[wv][(r + 8 * g) * 32 + nt * 16 + m] = (_Float16)v;
    }
  }
  __syncthreads();
  for (int nt = 0; nt < 2; ++nt) {
    v8f c = {};
    v16h a  = load_frag_a(&Hh[wv][0], 32, lane);
    v16h bb = load_frag_b_swz(W2s, 2, 0, nt, lane);
    c = WMMA_F32_F16(a, bb, c);
    const float bn = bias2[nt * 16 + m];
#pragma unroll
    for (int r = 0; r < 8; ++r)
      h2s[wv][(r + 8 * g) * 32 + nt * 16 + m] = c[r] + bn;
  }
  __syncthreads();
  const int r = lane >> 1, o = lane & 1;   // 16 rows x 2 logits per wave
  float acc = fb2[o];
  for (int c2 = 0; c2 < 32; ++c2) acc += h2s[wv][r * 32 + c2] * fw[c2 * 2 + o];
  out[(R0 + r) * 2 + o] = 1.f / (1.f + __expf(-acc));
}

// ---------------------------------------------------------------------------
extern "C" void kernel_launch(void* const* d_in, const int* in_sizes, int n_in,
                              void* d_out, int out_size, void* d_ws, size_t ws_size,
                              hipStream_t stream) {
  const float* x       = (const float*)d_in[0];
  const int*   sidx1   = (const int*)d_in[1];
  const int*   sidx2   = (const int*)d_in[2];
  const float* sa1_W1  = (const float*)d_in[3];
  const float* sa1_b1  = (const float*)d_in[4];
  const float* sa1_W2  = (const float*)d_in[5];
  const float* sa1_b2  = (const float*)d_in[6];
  const float* sa2_W1  = (const float*)d_in[7];
  const float* sa2_b1  = (const float*)d_in[8];
  const float* sa2_W2  = (const float*)d_in[9];
  const float* sa2_b2  = (const float*)d_in[10];
  const float* fp1_W1  = (const float*)d_in[11];
  const float* fp1_b1  = (const float*)d_in[12];
  const float* fp1_W2  = (const float*)d_in[13];
  const float* fp1_b2  = (const float*)d_in[14];
  const float* fp2_W1  = (const float*)d_in[15];
  const float* fp2_b1  = (const float*)d_in[16];
  const float* fp2_W2  = (const float*)d_in[17];
  const float* fp2_b2  = (const float*)d_in[18];
  const float* fc_W    = (const float*)d_in[19];
  const float* fc_b    = (const float*)d_in[20];
  float* out = (float*)d_out;

  char* ws = (char*)d_ws;
  size_t off = 0;
  auto alloc = [&](size_t bytes) -> void* {
    void* p = ws + off;
    off = (off + bytes + 255) & ~(size_t)255;
    return p;
  };
  float* samp1   = (float*)alloc((size_t)BB * S1 * 3 * 4);
  int*   idx1    = (int*)  alloc((size_t)BB * S1 * GG * 4);
  float* feat1   = (float*)alloc((size_t)BB * S1 * 64 * 4);
  int*   idx2    = (int*)  alloc((size_t)BB * S2 * GG * 4);
  float* feat2   = (float*)alloc((size_t)BB * S2 * 128 * 4);
  float* interp1 = (float*)alloc((size_t)BB * S1 * 128 * 4);
  float* h1      = (float*)alloc((size_t)BB * S1 * 64 * 4);
  float* interp2 = (float*)alloc((size_t)BB * NN * 64 * 4);
  (void)ws_size; (void)in_sizes; (void)n_in; (void)out_size;

  samp1_kernel<<<(BB * S1 + 255) / 256, 256, 0, stream>>>(x, sidx1, samp1);
  knn1_kernel<<<BB * S1, 32, 0, stream>>>(x, samp1, idx1);
  sa1_kernel<<<(BB * S1) / 4, 128, 0, stream>>>(x, idx1, sa1_W1, sa1_b1, sa1_W2, sa1_b2, feat1);
  knn2_kernel<<<BB * S2, 32, 0, stream>>>(feat1, sidx2, idx2);
  sa2_kernel<<<BB * S2, 32, 0, stream>>>(feat1, idx2, sa2_W1, sa2_b1, sa2_W2, sa2_b2, feat2);
  interp1_kernel<<<BB * 4, 128, 0, stream>>>(samp1, sidx2, feat2, interp1);
  fp1_kernel<<<(BB * S1) / 32, 64, 0, stream>>>(feat1, interp1, fp1_W1, fp1_b1, fp1_W2, fp1_b2, h1);
  interp2_kernel<<<BB * (NN / 256), 256, 0, stream>>>(x, samp1, h1, interp2);
  fp2_kernel<<<(BB * NN) / 64, 128, 0, stream>>>(x, interp2, fp2_W1, fp2_b1, fp2_W2, fp2_b2,
                                                 fc_W, fc_b, out);
}
